// OutputBlock_9388798509343
// MI455X (gfx1250) — compile-verified, matrix operand order checked by reading, not run on hardware
//
#include <hip/hip_runtime.h>
#include <hip/hip_bf16.h>
#include <cstdint>

// ---------------------------------------------------------------------------
// MI455X (gfx1250) fused GemNet OutputBlock.
// bf16 WMMA (v_wmma_f32_16x16x32_bf16) for the 7-GEMM MLP chains, fully fused
// per 48-row tile in LDS; fp32 accumulation; weights pre-transposed to bf16.
// ---------------------------------------------------------------------------

typedef __bf16 bf16;
typedef __attribute__((ext_vector_type(16))) __bf16 v16bf;
typedef __attribute__((ext_vector_type(8)))  __bf16 v8bf;
typedef __attribute__((ext_vector_type(8)))  float  v8f;

#define R_ROWS 48          // rows (edges/atoms) per block tile
#define LDK    264         // LDS row stride in bf16 elements (256 + 8 pad)
                           // pad=8 -> row-to-row bank stride of 4 banks:
                           // 16-lane b128 fragment loads are conflict-free.

__device__ __forceinline__ float scaled_silu(float x) {
  // GemNet ScaledSiLU: silu(x)/0.6
  return __fdividef(x, 1.0f + __expf(-x)) * 1.6666666666666667f;
}

__device__ __forceinline__ v16bf load_frag(const bf16* p) {
  // 16 bf16 per lane = two contiguous 16B chunks (K offsets {0..7},{16..23}
  // relative to the lane's hi*8 base) per the CDNA5 16-bit operand layout.
  union { v16bf v; v8bf h[2]; } u;
  u.h[0] = *(const v8bf*)(p);
  u.h[1] = *(const v8bf*)(p + 16);
  return u.v;
}

// One block = 256 threads = 8 waves. Computes OUT[R_ROWS][NDIM] =
// act(IN[R_ROWS][KDIM] @ W[KDIM][NDIM]) from LDS (IN) and global bf16 Wt
// (W transposed, [NDIM][KDIM]). If lds_res != nullptr, applies the GemNet
// residual combine: out = (res + silu(acc)) * (1/sqrt(2)), in place on res.
template<int KDIM, int NDIM>
__device__ __forceinline__ void gemm_block(const bf16* __restrict__ lds_in,
                                           const bf16* __restrict__ Wt,
                                           bf16* __restrict__ lds_out,
                                           const bf16* __restrict__ lds_res) {
  const int lane = threadIdx.x & 31;
  const int wave = threadIdx.x >> 5;
  const int n0   = lane & 15;   // row (A) / column (B,C) index within tile
  const int hi   = lane >> 4;   // which K half-group this lane holds
  constexpr int NT  = NDIM / 16;     // column tiles
  constexpr int RT  = R_ROWS / 16;   // row tiles (3)
  constexpr int CPW = NT / 8;        // column tiles per wave (1 or 2)

  for (int ci = 0; ci < CPW; ++ci) {
    const int ct = wave * CPW + ci;
    v8f acc[RT] = {};
    const bf16* wrow  = Wt + (size_t)(ct * 16 + n0) * KDIM + hi * 8;
    const bf16* arow0 = lds_in + n0 * LDK + hi * 8;
#pragma unroll
    for (int k0 = 0; k0 < KDIM; k0 += 32) {
      v16bf bfrag = load_frag(wrow + k0);
#pragma unroll
      for (int rt = 0; rt < RT; ++rt) {
        v16bf afrag = load_frag(arow0 + rt * 16 * LDK + k0);
        acc[rt] = __builtin_amdgcn_wmma_f32_16x16x32_bf16(
            false, afrag, false, bfrag, (short)0, acc[rt], false, false);
      }
    }
    // Epilogue: C/D layout -> lane holds rows (hi*8 + r), column n0.
#pragma unroll
    for (int rt = 0; rt < RT; ++rt) {
#pragma unroll
      for (int r = 0; r < 8; ++r) {
        const int row = rt * 16 + hi * 8 + r;
        const int col = ct * 16 + n0;
        float v = scaled_silu(acc[rt][r]);
        if (lds_res) {
          v = ((float)lds_res[row * LDK + col] + v) * 0.70710678118654752f;
        }
        lds_out[row * LDK + col] = (bf16)v;
      }
    }
  }
}

// --------------------------- weight preparation ----------------------------
// W[K][N] f32 (row-major) -> Wt[N][K] bf16, batched over blockIdx.y.
__global__ void k_transpose_bf16(const float* __restrict__ W,
                                 bf16* __restrict__ Wt, int K, int N) {
  const size_t mat = (size_t)K * N;
  W  += mat * blockIdx.y;
  Wt += mat * blockIdx.y;
  for (int idx = blockIdx.x * blockDim.x + threadIdx.x; idx < (int)mat;
       idx += gridDim.x * blockDim.x) {
    const int k = idx / N, n = idx - k * N;
    Wt[(size_t)n * K + k] = (bf16)W[idx];
  }
}

// ------------------------ energy branch: edge scatter ----------------------
// xE[id_j[e]] += m[e] * (rbf[e] @ We_rbf), coalesced f32 global atomics.
__global__ __launch_bounds__(256)
void k_energy_scatter(const float* __restrict__ m, const float* __restrict__ rbf,
                      const int* __restrict__ id_j,
                      const float* __restrict__ We_rbf,
                      float* __restrict__ xE, int nE) {
  __shared__ float Wr[16 * 256];
  __shared__ float rb[4][16];
  for (int i = threadIdx.x; i < 16 * 256; i += 256) Wr[i] = We_rbf[i];
  const int e0 = blockIdx.x * 4;
  if (threadIdx.x < 64) {
    const int e = e0 + (threadIdx.x >> 4), k = threadIdx.x & 15;
    rb[threadIdx.x >> 4][k] = (e < nE) ? rbf[(size_t)e * 16 + k] : 0.0f;
  }
  __syncthreads();
  const int c = threadIdx.x;
#pragma unroll
  for (int i = 0; i < 4; ++i) {
    const int e = e0 + i;
    if (e >= nE) break;
    float acc = 0.0f;
#pragma unroll
    for (int k = 0; k < 16; ++k) acc += rb[i][k] * Wr[k * 256 + c];
    acc *= m[(size_t)e * 256 + c];
    atomicAdd(&xE[(size_t)id_j[e] * 256 + c], acc);
  }
}

// --------------------------- fused energy MLP ------------------------------
__global__ __launch_bounds__(256)
void k_energy_mlp(const float* __restrict__ xE, const bf16* __restrict__ WtIn,
                  const bf16* __restrict__ WtRes, const float* __restrict__ Wout,
                  float* __restrict__ E, int nA) {
  __shared__ bf16 bufA[R_ROWS * LDK];
  __shared__ bf16 bufB[R_ROWS * LDK];
  const int r0 = blockIdx.x * R_ROWS;
  for (int idx = threadIdx.x; idx < R_ROWS * 256; idx += 256) {
    const int r = idx >> 8, c = idx & 255;
    const float v = (r0 + r < nA) ? xE[(size_t)(r0 + r) * 256 + c] : 0.0f;
    bufA[r * LDK + c] = (bf16)v;
  }
  __syncthreads();
  gemm_block<256, 128>(bufA, WtIn, bufB, nullptr);       // dense 256->128
  __syncthreads();
  for (int h = 0; h < 3; ++h) {                          // residual stack
    gemm_block<128, 128>(bufB, WtRes + (size_t)(2 * h) * 128 * 128, bufA, nullptr);
    __syncthreads();
    gemm_block<128, 128>(bufA, WtRes + (size_t)(2 * h + 1) * 128 * 128, bufB, bufB);
    __syncthreads();
  }
  // E = x @ We_out : 4 lanes per row, 32 columns each, shuffle-reduce.
  const int row = threadIdx.x >> 2, part = threadIdx.x & 3;
  float partial = 0.0f;
  const bool live = (row < R_ROWS) && (r0 + row < nA);
  if (live) {
    for (int c = part * 32; c < part * 32 + 32; ++c)
      partial += (float)bufB[row * LDK + c] * Wout[c];
  }
  partial += __shfl_xor(partial, 1);
  partial += __shfl_xor(partial, 2);
  if (part == 0 && live) E[r0 + row] = partial;
}

// ---------------------------- fused force MLP ------------------------------
__global__ __launch_bounds__(256)
void k_force_mlp(const float* __restrict__ m, const float* __restrict__ rbf,
                 const bf16* __restrict__ WtIn, const bf16* __restrict__ WtRes,
                 const float* __restrict__ Wrbf, const float* __restrict__ Wout,
                 float* __restrict__ F, int nE) {
  __shared__ bf16 bufA[R_ROWS * LDK];
  __shared__ bf16 bufB[R_ROWS * LDK];
  const int e0 = blockIdx.x * R_ROWS;
  for (int idx = threadIdx.x; idx < R_ROWS * 256; idx += 256) {
    const int r = idx >> 8, c = idx & 255;
    const float v = (e0 + r < nE) ? m[(size_t)(e0 + r) * 256 + c] : 0.0f;
    bufA[r * LDK + c] = (bf16)v;
  }
  __syncthreads();
  gemm_block<256, 256>(bufA, WtIn, bufB, nullptr);       // dense 256->256
  __syncthreads();
  for (int h = 0; h < 3; ++h) {                          // residual stack
    gemm_block<256, 256>(bufB, WtRes + (size_t)(2 * h) * 256 * 256, bufA, nullptr);
    __syncthreads();
    gemm_block<256, 256>(bufA, WtRes + (size_t)(2 * h + 1) * 256 * 256, bufB, bufB);
    __syncthreads();
  }
  // F = (x * (rbf @ Wf_rbf)) @ Wf_out : 4 lanes/row, 64 columns each.
  const int row = threadIdx.x >> 2, part = threadIdx.x & 3;
  const int e = e0 + row;
  float partial = 0.0f;
  const bool live = (row < R_ROWS) && (e < nE);
  if (live) {
    float rb[16];
#pragma unroll
    for (int k = 0; k < 16; ++k) rb[k] = rbf[(size_t)e * 16 + k];
    for (int c = part * 64; c < part * 64 + 64; ++c) {
      float rf = 0.0f;
#pragma unroll
      for (int k = 0; k < 16; ++k) rf += rb[k] * Wrbf[k * 256 + c];
      partial += (float)bufB[row * LDK + c] * rf * Wout[c];
    }
  }
  partial += __shfl_xor(partial, 1);
  partial += __shfl_xor(partial, 2);
  if (part == 0 && live) F[e] = partial;
}

// ---------------------------------------------------------------------------
extern "C" void kernel_launch(void* const* d_in, const int* in_sizes, int n_in,
                              void* d_out, int out_size, void* d_ws, size_t ws_size,
                              hipStream_t stream) {
  (void)n_in; (void)out_size; (void)ws_size;
  const float* m      = (const float*)d_in[1];
  const float* rbf    = (const float*)d_in[2];
  const int*   id_j   = (const int*)d_in[3];
  const float* We_rbf = (const float*)d_in[4];
  const float* We_in  = (const float*)d_in[5];
  const float* We_res = (const float*)d_in[6];
  const float* We_out = (const float*)d_in[7];
  const float* Wf_rbf = (const float*)d_in[8];
  const float* Wf_in  = (const float*)d_in[9];
  const float* Wf_res = (const float*)d_in[10];
  const float* Wf_out = (const float*)d_in[11];

  const int dA = 128, dE = 256;
  const int nA = in_sizes[0] / dA;   // 25000
  const int nE = in_sizes[1] / dE;   // 400000

  float* E = (float*)d_out;
  float* F = E + nA;

  // Workspace carve-up (256B aligned).
  char*  ws  = (char*)d_ws;
  size_t off = 0;
  auto carve = [&](size_t bytes) {
    void* p = ws + off;
    off = (off + bytes + 255) & ~(size_t)255;
    return p;
  };
  float* xE     = (float*)carve((size_t)nA * dE * sizeof(float));
  bf16*  WtEin  = (bf16*)carve((size_t)dA * dE * sizeof(bf16));
  bf16*  WtEres = (bf16*)carve((size_t)6 * dA * dA * sizeof(bf16));
  bf16*  WtFin  = (bf16*)carve((size_t)dE * dE * sizeof(bf16));
  bf16*  WtFres = (bf16*)carve((size_t)6 * dE * dE * sizeof(bf16));

  hipMemsetAsync(xE, 0, (size_t)nA * dE * sizeof(float), stream);

  // Weight prep: f32 -> transposed bf16 (tiny; L2-resident thereafter).
  k_transpose_bf16<<<dim3((dE * dA + 255) / 256, 1), 256, 0, stream>>>(We_in, WtEin, dE, dA);
  k_transpose_bf16<<<dim3((dA * dA + 255) / 256, 6), 256, 0, stream>>>(We_res, WtEres, dA, dA);
  k_transpose_bf16<<<dim3((dE * dE + 255) / 256, 1), 256, 0, stream>>>(Wf_in, WtFin, dE, dE);
  k_transpose_bf16<<<dim3((dE * dE + 255) / 256, 6), 256, 0, stream>>>(Wf_res, WtFres, dE, dE);

  // Energy branch.
  k_energy_scatter<<<(nE + 3) / 4, 256, 0, stream>>>(m, rbf, id_j, We_rbf, xE, nE);
  k_energy_mlp<<<(nA + R_ROWS - 1) / R_ROWS, 256, 0, stream>>>(xE, WtEin, WtEres, We_out, E, nA);

  // Force branch (dominant cost; fused bf16-WMMA GEMM chain).
  k_force_mlp<<<(nE + R_ROWS - 1) / R_ROWS, 256, 0, stream>>>(m, rbf, WtFin, WtFres, Wf_rbf, Wf_out, F, nE);
}
